// HybridModel_33397665694047
// MI455X (gfx1250) — matrix operand order repaired and checked by reading, not compiled, not run on hardware
//
#include <hip/hip_runtime.h>
#include <hip/hip_bf16.h>
#include <math.h>

// ---------------------------------------------------------------------------
// 3-NN soft interpolation on MI455X (gfx1250, wave32, WMMA).
//   dp   = source_feats @ target_feats^T        (bf16x3 split -> ~fp32 acc)
//   dist = t_sq[j] - 2*dp   (s_sq dropped: shift-invariant for topk+softmax)
// Top-3 kept as packed f64 keys: bitcast(double(dist)) | index (low 29
// mantissa bits of a f32->f64 widen are zero). All-positive keys => f64
// minnum/maxnum ordering == (dist, index) lexicographic, ties to smaller
// index (matches lax.top_k). Insert = branch-free 5-op min/max network in
// raw v_min_num_f64/v_max_num_f64 inline asm (no canonicalize padding).
// B tiles are ping-pong double-buffered with one-tile lookahead so WMMAs
// never wait on just-issued loads. Target arrays padded to a multiple of
// 32 rows (+1 extra zero tile for the lookahead overshoot); pad t_sq=1e30
// so padded columns can never win -> no masking in the hot loop.
// ---------------------------------------------------------------------------

typedef __bf16 bf16_t;
typedef bf16_t v16bf __attribute__((ext_vector_type(16)));
typedef float  v8f   __attribute__((ext_vector_type(8)));

#define FEAT_D 128
#define NCHUNK 4                  // 4 chunks of K=32
#define BT     (16 * FEAT_D)      // elements per 16-row B tile
#define PAD_DIST 1e30f
#define IDX_MASK 0x1FFFFFFFull    // 29 zero bits below a widened f32 mantissa

__device__ __forceinline__ double dmin(double a, double b) {
  double r;
  asm("v_min_num_f64 %0, %1, %2" : "=v"(r) : "v"(a), "v"(b));
  return r;
}
__device__ __forceinline__ double dmax(double a, double b) {
  double r;
  asm("v_max_num_f64 %0, %1, %2" : "=v"(r) : "v"(a), "v"(b));
  return r;
}

__device__ __forceinline__ double mk_key(float d, int j) {
  unsigned long long b = __builtin_bit_cast(unsigned long long, (double)d);
  b |= (unsigned int)j;
  return __builtin_bit_cast(double, b);
}

// Branch-free sorted insert of key into (k0 <= k1 <= k2).
__device__ __forceinline__ void ins3(double key, double& k0, double& k1, double& k2) {
  double u0 = dmax(k0, key);
  k0 = dmin(k0, key);
  double u1 = dmax(k1, u0);
  k1 = dmin(k1, u0);
  k2 = dmin(k2, u1);
}

__device__ __forceinline__ void load_btile(const bf16_t* __restrict__ bh_p,
                                           const bf16_t* __restrict__ bl_p,
                                           v16bf bh[NCHUNK], v16bf bl[NCHUNK]) {
  #pragma unroll
  for (int ch = 0; ch < NCHUNK; ++ch) {
    bh[ch] = *(const v16bf*)(bh_p + 32 * ch);
    bl[ch] = *(const v16bf*)(bl_p + 32 * ch);
  }
}

__device__ __forceinline__ void do_tile(const v16bf ahi[NCHUNK], const v16bf alo[NCHUNK],
                                        const v16bf bh[NCHUNK], const v16bf bl[NCHUNK],
                                        float tq, int jcol,
                                        double k0a[8], double k1a[8], double k2a[8]) {
  v8f acc = {0.f,0.f,0.f,0.f,0.f,0.f,0.f,0.f};
  #pragma unroll
  for (int ch = 0; ch < NCHUNK; ++ch) {
    acc = __builtin_amdgcn_wmma_f32_16x16x32_bf16(false, ahi[ch], false, bh[ch],
                                                  (short)0, acc, false, false);
    acc = __builtin_amdgcn_wmma_f32_16x16x32_bf16(false, alo[ch], false, bh[ch],
                                                  (short)0, acc, false, false);
    acc = __builtin_amdgcn_wmma_f32_16x16x32_bf16(false, ahi[ch], false, bl[ch],
                                                  (short)0, acc, false, false);
  }
  #pragma unroll
  for (int v = 0; v < 8; ++v) {
    const float d = fmaf(-2.f, acc[v], tq);
    ins3(mk_key(d, jcol), k0a[v], k1a[v], k2a[v]);
  }
}

// Split f32 -> (bf16 hi, bf16 lo) so hi+lo ~= x to ~16 mantissa bits.
// Writes npad elements; elements >= n are zero (tile padding).
__global__ void split_hi_lo_kernel(const float* __restrict__ src,
                                   bf16_t* __restrict__ hi,
                                   bf16_t* __restrict__ lo,
                                   int n, int npad) {
  int i = blockIdx.x * blockDim.x + threadIdx.x;
  if (i < npad) {
    float x = (i < n) ? src[i] : 0.f;
    bf16_t h = (bf16_t)x;
    hi[i] = h;
    lo[i] = (bf16_t)(x - (float)h);
  }
}

// Per-target-row squared norm (exact f32); pad rows get PAD_DIST.
__global__ void tsq_kernel(const float* __restrict__ tf,
                           float* __restrict__ tsq, int M, int Mext) {
  int j = blockIdx.x * blockDim.x + threadIdx.x;
  if (j < Mext) {
    float s = PAD_DIST;
    if (j < M) {
      const float4* p = (const float4*)(tf + (size_t)j * FEAT_D);
      s = 0.f;
      #pragma unroll
      for (int t = 0; t < FEAT_D / 4; ++t) {
        float4 v = p[t];
        s = fmaf(v.x, v.x, s); s = fmaf(v.y, v.y, s);
        s = fmaf(v.z, v.z, s); s = fmaf(v.w, v.w, s);
      }
    }
    tsq[j] = s;
  }
}

__launch_bounds__(64)
__global__ void knn_top3_kernel(const float*  __restrict__ sf,   // [N,128]
                                const bf16_t* __restrict__ thi,  // [Mext,128]
                                const bf16_t* __restrict__ tlo,  // [Mext,128]
                                const float*  __restrict__ tsq,  // [Mext]
                                const float*  __restrict__ tp,   // [M,3]
                                float*        __restrict__ out,  // [N,3]
                                int N, int Mpad) {
  __shared__ double s_k[2][16][16][3];     // packed (dist|idx) keys, 12 KB

  const int wave = threadIdx.x >> 5;
  const int lane = threadIdx.x & 31;
  const int h    = lane >> 4;      // which 16-lane half
  const int c    = lane & 15;
  const int row0 = (blockIdx.x * 2 + wave) * 16;

  // ---- Preload A tile (16 rows x 128) into bf16 hi/lo WMMA fragments.
  // A 16x32 fragment order used here (same as B): lane row = c; element e -> K = 16h + e.
  v16bf ahi[NCHUNK], alo[NCHUNK];
  {
    int arow = row0 + c; if (arow > N - 1) arow = N - 1;
    const float* ap = sf + (size_t)arow * FEAT_D;
    #pragma unroll
    for (int ch = 0; ch < NCHUNK; ++ch) {
      const int k0 = 32 * ch;
      float f[16];
      const float4* p0 = (const float4*)(ap + k0 + 16 * h);
      float4 q0 = p0[0], q1 = p0[1];
      const float4* p1 = (const float4*)(ap + k0 + 16 * h + 8);
      float4 q2 = p1[0], q3 = p1[1];
      f[0]=q0.x;  f[1]=q0.y;  f[2]=q0.z;  f[3]=q0.w;
      f[4]=q1.x;  f[5]=q1.y;  f[6]=q1.z;  f[7]=q1.w;
      f[8]=q2.x;  f[9]=q2.y;  f[10]=q2.z; f[11]=q2.w;
      f[12]=q3.x; f[13]=q3.y; f[14]=q3.z; f[15]=q3.w;
      #pragma unroll
      for (int e = 0; e < 16; ++e) {
        bf16_t hv = (bf16_t)f[e];
        ahi[ch][e] = hv;
        alo[ch][e] = (bf16_t)(f[e] - (float)hv);
      }
    }
  }

  const double DINF = __builtin_inf();
  double k0a[8], k1a[8], k2a[8];
  #pragma unroll
  for (int v = 0; v < 8; ++v) { k0a[v] = k1a[v] = k2a[v] = DINF; }

  // ---- Sweep all column tiles of 16 target rows, 2 tiles per iteration.
  // B 32x16 fragment order used here (same as A): lane col = c; element e -> K = 16h + e.
  const bf16_t* bh_p = thi + (size_t)c * FEAT_D + 16 * h;
  const bf16_t* bl_p = tlo + (size_t)c * FEAT_D + 16 * h;
  const float*  tq_p = tsq + c;
  int jcol = c;

  v16bf bhA[NCHUNK], blA[NCHUNK], bhB[NCHUNK], blB[NCHUNK];
  load_btile(bh_p, bl_p, bhA, blA);               // tile 0

  const int numJT = Mpad >> 4;                    // even (Mpad % 32 == 0)
  for (int jt = 0; jt < numJT; jt += 2) {
    load_btile(bh_p + BT, bl_p + BT, bhB, blB);   // tile jt+1
    do_tile(ahi, alo, bhA, blA, tq_p[0], jcol, k0a, k1a, k2a);
    load_btile(bh_p + 2 * BT, bl_p + 2 * BT, bhA, blA);  // tile jt+2 (pad-safe)
    do_tile(ahi, alo, bhB, blB, tq_p[16], jcol + 16, k0a, k1a, k2a);
    bh_p += 2 * BT;
    bl_p += 2 * BT;
    tq_p += 32;
    jcol += 32;
  }

  // ---- Stage per-lane top-3 keys, merge the 16 column-lanes per row.
  #pragma unroll
  for (int v = 0; v < 8; ++v) {
    const int r = v + 8 * h;
    s_k[wave][r][c][0] = k0a[v];
    s_k[wave][r][c][1] = k1a[v];
    s_k[wave][r][c][2] = k2a[v];
  }
  __syncthreads();

  if (lane < 16) {
    const int r = lane;
    double e0 = DINF, e1 = DINF, e2 = DINF;
    for (int cc = 0; cc < 16; ++cc) {
      #pragma unroll
      for (int t = 0; t < 3; ++t) {
        ins3(s_k[wave][r][cc][t], e0, e1, e2);
      }
    }
    const int orow = row0 + r;
    if (orow < N) {
      unsigned long long q0 = __builtin_bit_cast(unsigned long long, e0);
      unsigned long long q1 = __builtin_bit_cast(unsigned long long, e1);
      unsigned long long q2 = __builtin_bit_cast(unsigned long long, e2);
      const int m0 = (int)(q0 & IDX_MASK);
      const int m1 = (int)(q1 & IDX_MASK);
      const int m2 = (int)(q2 & IDX_MASK);
      const float d0 = (float)__builtin_bit_cast(double, q0 & ~IDX_MASK);
      const float d1 = (float)__builtin_bit_cast(double, q1 & ~IDX_MASK);
      const float d2 = (float)__builtin_bit_cast(double, q2 & ~IDX_MASK);
      // softmax over scores = -dist; shift by -d0 (largest score).
      float w0 = 1.f;
      float w1 = __expf(d0 - d1);
      float w2 = __expf(d0 - d2);
      float inv = 1.f / (w0 + w1 + w2);
      w0 *= inv; w1 *= inv; w2 *= inv;
      float px = tp[m0*3+0]*w0 + tp[m1*3+0]*w1 + tp[m2*3+0]*w2;
      float py = tp[m0*3+1]*w0 + tp[m1*3+1]*w1 + tp[m2*3+1]*w2;
      float pz = tp[m0*3+2]*w0 + tp[m1*3+2]*w1 + tp[m2*3+2]*w2;
      out[orow*3+0] = px; out[orow*3+1] = py; out[orow*3+2] = pz;
    }
  }
}

extern "C" void kernel_launch(void* const* d_in, const int* in_sizes, int n_in,
                              void* d_out, int out_size, void* d_ws, size_t ws_size,
                              hipStream_t stream) {
  const float* sf = (const float*)d_in[0];   // source_feats [N,128]
  const float* tf = (const float*)d_in[1];   // target_feats [M,128]
  const float* tp = (const float*)d_in[2];   // target_points [M,3]
  const int N    = in_sizes[0] / FEAT_D;     // 100000
  const int M    = in_sizes[1] / FEAT_D;     // 6890
  const int Mpad = ((M + 31) / 32) * 32;     // 6912 (even tile count)
  const int Mext = Mpad + 16;                // +1 zero tile for lookahead

  // Workspace: [thi: Mext*128 bf16][tlo: Mext*128 bf16][tsq: Mext f32] (~3.6MB)
  bf16_t* thi = (bf16_t*)d_ws;
  bf16_t* tlo = thi + (size_t)Mext * FEAT_D;
  float*  tsq = (float*)(tlo + (size_t)Mext * FEAT_D);

  const int nmd  = M * FEAT_D;
  const int next = Mext * FEAT_D;
  split_hi_lo_kernel<<<(next + 255) / 256, 256, 0, stream>>>(tf, thi, tlo, nmd, next);
  tsq_kernel<<<(Mext + 255) / 256, 256, 0, stream>>>(tf, tsq, M, Mext);

  const int tiles  = (N + 15) / 16;          // 6250 row tiles
  const int blocks = (tiles + 1) / 2;        // 2 waves (tiles) per block
  knn_top3_kernel<<<blocks, 64, 0, stream>>>(sf, thi, tlo, tsq, tp,
                                             (float*)d_out, N, Mpad);
}
